// UniversalHiCGNN_7043746365844
// MI455X (gfx1250) — matrix operand order repaired and checked by reading, not compiled, not run on hardware
//
#include <hip/hip_runtime.h>

#define NNODES 8192
#define HDIM   128

typedef __attribute__((ext_vector_type(16))) __bf16 v16bf;
typedef __attribute__((ext_vector_type(8)))  float  v8f;

// ---------- bf16 helpers (RTNE via compiler fptrunc) ----------
static __device__ inline unsigned short f2bf(float f) {
    __bf16 h = (__bf16)f;
    return __builtin_bit_cast(unsigned short, h);
}
static __device__ inline float bf2f(unsigned short u) {
    __bf16 h = __builtin_bit_cast(__bf16, u);
    return (float)h;
}
static __device__ inline float bf_round(float f) { return bf2f(f2bf(f)); }
static __device__ inline unsigned pack2(float a, float b) {
    return (unsigned)f2bf(a) | ((unsigned)f2bf(b) << 16);
}

// ---------- WMMA fragment loaders (layouts per CDNA5 ISA 7.12.2) ----------
// Each lane's fragment is 32 LDS bytes in two 16-byte runs 16 bytes apart -> two aligned b128 loads.
union FragBF { v16bf v; uint4 q[2]; unsigned u[8]; };

// A matrix 16x32 bf16 from LDS tile row-major [rows][32 bf16] (16 u32/row).
// lane: m = lane&15 (+mBase); kh = lane>>4.
// u[v<4] = row[kh*4 + v], u[v>=4] = row[8 + kh*4 + (v-4)]  (k-pairs packed).
static __device__ inline v16bf load_a(const unsigned* s, int mBase, int lane) {
    const unsigned* row = s + (mBase + (lane & 15)) * 16 + (lane >> 4) * 4;
    FragBF f;
    f.q[0] = *(const uint4*)(row);      // 16B aligned
    f.q[1] = *(const uint4*)(row + 8);
    return f.v;
}

// B matrix 32x16 bf16 from LDS tile stored transposed [n][k] (16 u32/row).
// lane: n = lane&15 (+nBase); kh = lane>>4; u[v<4] = row[kh*4 + v], u[v>=4] = row[8 + kh*4 + (v-4)].
static __device__ inline v16bf load_b(const unsigned* s, int nBase, int lane) {
    const unsigned* row = s + (nBase + (lane & 15)) * 16 + (lane >> 4) * 4;
    FragBF f;
    f.q[0] = *(const uint4*)(row);
    f.q[1] = *(const uint4*)(row + 8);
    return f.v;
}

static __device__ inline v8f wmma_bf16(v16bf a, v16bf b, v8f c) {
    return __builtin_amdgcn_wmma_f32_16x16x32_bf16(
        /*neg_a=*/false, a, /*neg_b=*/false, b,
        /*c_mod=*/(short)0, c, /*reuse_a=*/false, /*reuse_b=*/false);
}

// ---------------------------------------------------------------------------
// deg[i] = max(sum_j adj[i,j], 1)
// ---------------------------------------------------------------------------
__global__ __launch_bounds__(256)
void deg_kernel(float* __restrict__ deg, const float* __restrict__ adj) {
    __shared__ float red[256];
    const int row = blockIdx.x;
    const float* p = adj + (size_t)row * NNODES;
    float s = 0.f;
    for (int c = threadIdx.x * 4; c < NNODES; c += 256 * 4) {
        const float4 v = *(const float4*)(p + c);
        s += v.x + v.y + v.z + v.w;
    }
    red[threadIdx.x] = s;
    __syncthreads();
    for (int st = 128; st > 0; st >>= 1) {
        if (threadIdx.x < st) red[threadIdx.x] += red[threadIdx.x + st];
        __syncthreads();
    }
    if (threadIdx.x == 0) deg[row] = fmaxf(red[0], 1.0f);
}

// ---------------------------------------------------------------------------
// C[M,F] = (adj @ X) / deg.  adj exact in bf16 (0/1); X split hi/lo bf16.
// Block 256 thr (8 waves): tile M=128 (16 rows/wave), Ftile=128, K-step 32.
// Double-buffered LDS with register staging: next tile's global loads are
// issued before the WMMA burst so HBM streaming overlaps the matrix pipe.
// ---------------------------------------------------------------------------
__global__ __launch_bounds__(256)
void agg_kernel(float* __restrict__ C, const float* __restrict__ adj,
                const float* __restrict__ X, const float* __restrict__ deg,
                int F) {
    __shared__ unsigned s_adj[2][128 * 16];   // rows x 32 bf16 (row-major)
    __shared__ unsigned s_xhi[2][128 * 16];   // feats x 32 nodes ([n][k])
    __shared__ unsigned s_xlo[2][128 * 16];

    const int tid   = threadIdx.x;
    const int lane  = tid & 31;
    const int wave  = tid >> 5;                 // 0..7
    const int mBase = blockIdx.y * 128;
    const int f0    = blockIdx.x * 128;

    const v8f zero = {0.f, 0.f, 0.f, 0.f, 0.f, 0.f, 0.f, 0.f};
    v8f acc[8];
#pragma unroll
    for (int t = 0; t < 8; ++t) acc[t] = zero;

    float4 sa[4];            // staged adj tile slice (16 f32/thread)
    float  sxa[8], sxb[8];   // staged X tile slice (16 f32/thread)

    auto load_stage = [&](int k0) {
#pragma unroll
        for (int j = 0; j < 4; ++j) {
            const int idx4 = tid + j * 256;     // 0..1023
            const int r = idx4 >> 3;            // row 0..127
            const int c = (idx4 & 7) << 2;      // col 0,4,..,28
            sa[j] = *(const float4*)(adj + (size_t)(mBase + r) * NNODES + k0 + c);
        }
#pragma unroll
        for (int j = 0; j < 8; ++j) {
            const int u  = tid + j * 256;       // 0..2047
            const int kp = u >> 7;              // k-pair 0..15
            const int n  = u & 127;
            sxa[j] = X[(size_t)(k0 + 2 * kp)     * F + f0 + n];
            sxb[j] = X[(size_t)(k0 + 2 * kp + 1) * F + f0 + n];
        }
    };
    auto store_stage = [&](int buf) {
#pragma unroll
        for (int j = 0; j < 4; ++j) {
            const int idx4 = tid + j * 256;
            const int r = idx4 >> 3;
            const int c = (idx4 & 7) << 2;
            s_adj[buf][r * 16 + (c >> 1)]     = pack2(sa[j].x, sa[j].y);
            s_adj[buf][r * 16 + (c >> 1) + 1] = pack2(sa[j].z, sa[j].w);
        }
#pragma unroll
        for (int j = 0; j < 8; ++j) {
            const int u  = tid + j * 256;
            const int kp = u >> 7;
            const int n  = u & 127;
            s_xhi[buf][n * 16 + kp] = pack2(sxa[j], sxb[j]);
            s_xlo[buf][n * 16 + kp] = pack2(sxa[j] - bf_round(sxa[j]),
                                            sxb[j] - bf_round(sxb[j]));
        }
    };

    load_stage(0);
    store_stage(0);
    __syncthreads();

    const int STEPS = NNODES / 32;
    for (int s = 0; s < STEPS; ++s) {
        const int cur = s & 1;
        if (s + 1 < STEPS) load_stage((s + 1) * 32);   // issue HBM loads first

        const v16bf af = load_a(s_adj[cur], wave * 16, lane);
#pragma unroll
        for (int t = 0; t < 8; ++t) {
            acc[t] = wmma_bf16(af, load_b(s_xhi[cur], t * 16, lane), acc[t]);
            acc[t] = wmma_bf16(af, load_b(s_xlo[cur], t * 16, lane), acc[t]);
        }

        if (s + 1 < STEPS) store_stage(cur ^ 1);       // convert + fill other buf
        __syncthreads();
    }

    // epilogue: divide by degree, store f32
    const int nLoc = lane & 15;
    const int mh   = lane >> 4;
#pragma unroll
    for (int v = 0; v < 8; ++v) {
        const int m = mBase + wave * 16 + mh * 8 + v;
        const float inv = 1.0f / deg[m];
        float* crow = C + (size_t)m * F + f0;
#pragma unroll
        for (int t = 0; t < 8; ++t) crow[t * 16 + nLoc] = acc[t][v] * inv;
    }
}

// ---------------------------------------------------------------------------
// Hout[M,128] = act(A1@W1 + A2@W2 + b); hi/lo split on both operands.
// Block 256 thr: tile M=128, Ntile=64, K-step 32.
// ---------------------------------------------------------------------------
__global__ __launch_bounds__(256)
void lin_kernel(float* __restrict__ Hout,
                const float* __restrict__ A1, const float* __restrict__ A2,
                const float* __restrict__ W1, const float* __restrict__ W2,
                const float* __restrict__ b, int K, int relu) {
    __shared__ unsigned s_a1h[128 * 16], s_a1l[128 * 16];
    __shared__ unsigned s_a2h[128 * 16], s_a2l[128 * 16];
    __shared__ unsigned s_w1h[64 * 16],  s_w1l[64 * 16];
    __shared__ unsigned s_w2h[64 * 16],  s_w2l[64 * 16];

    const int tid   = threadIdx.x;
    const int lane  = tid & 31;
    const int wave  = tid >> 5;
    const int mBase = blockIdx.y * 128;
    const int f0    = blockIdx.x * 64;

    const v8f zero = {0.f, 0.f, 0.f, 0.f, 0.f, 0.f, 0.f, 0.f};
    v8f acc[4];
#pragma unroll
    for (int t = 0; t < 4; ++t) acc[t] = zero;

    for (int k0 = 0; k0 < K; k0 += 32) {
        // A tiles 128x32 (row-major, pairs contiguous along k)
#pragma unroll
        for (int j = 0; j < 4; ++j) {
            const int idx4 = tid + j * 256;
            const int r = idx4 >> 3;
            const int c = (idx4 & 7) << 2;
            const int so = r * 16 + (c >> 1);
            const float4 v1 = *(const float4*)(A1 + (size_t)(mBase + r) * K + k0 + c);
            s_a1h[so]     = pack2(v1.x, v1.y);
            s_a1h[so + 1] = pack2(v1.z, v1.w);
            s_a1l[so]     = pack2(v1.x - bf_round(v1.x), v1.y - bf_round(v1.y));
            s_a1l[so + 1] = pack2(v1.z - bf_round(v1.z), v1.w - bf_round(v1.w));
            const float4 v2 = *(const float4*)(A2 + (size_t)(mBase + r) * K + k0 + c);
            s_a2h[so]     = pack2(v2.x, v2.y);
            s_a2h[so + 1] = pack2(v2.z, v2.w);
            s_a2l[so]     = pack2(v2.x - bf_round(v2.x), v2.y - bf_round(v2.y));
            s_a2l[so + 1] = pack2(v2.z - bf_round(v2.z), v2.w - bf_round(v2.w));
        }
        // W tiles 32x64 -> transposed [n][k] (1024 u32 pairs, 4 per thread)
#pragma unroll
        for (int j = 0; j < 4; ++j) {
            const int u  = tid + j * 256;           // 0..1023
            const int kp = u >> 6;                  // 0..15
            const int n  = u & 63;
            const int so = n * 16 + kp;
            const float a1 = W1[(size_t)(k0 + 2 * kp)     * HDIM + f0 + n];
            const float b1 = W1[(size_t)(k0 + 2 * kp + 1) * HDIM + f0 + n];
            s_w1h[so] = pack2(a1, b1);
            s_w1l[so] = pack2(a1 - bf_round(a1), b1 - bf_round(b1));
            const float a2 = W2[(size_t)(k0 + 2 * kp)     * HDIM + f0 + n];
            const float b2 = W2[(size_t)(k0 + 2 * kp + 1) * HDIM + f0 + n];
            s_w2h[so] = pack2(a2, b2);
            s_w2l[so] = pack2(a2 - bf_round(a2), b2 - bf_round(b2));
        }
        __syncthreads();

        const v16bf a1h = load_a(s_a1h, wave * 16, lane);
        const v16bf a1l = load_a(s_a1l, wave * 16, lane);
        const v16bf a2h = load_a(s_a2h, wave * 16, lane);
        const v16bf a2l = load_a(s_a2l, wave * 16, lane);
#pragma unroll
        for (int t = 0; t < 4; ++t) {
            const v16bf b1h = load_b(s_w1h, t * 16, lane);
            const v16bf b1l = load_b(s_w1l, t * 16, lane);
            acc[t] = wmma_bf16(a1h, b1h, acc[t]);
            acc[t] = wmma_bf16(a1h, b1l, acc[t]);
            acc[t] = wmma_bf16(a1l, b1h, acc[t]);
            acc[t] = wmma_bf16(a1l, b1l, acc[t]);
            const v16bf b2h = load_b(s_w2h, t * 16, lane);
            const v16bf b2l = load_b(s_w2l, t * 16, lane);
            acc[t] = wmma_bf16(a2h, b2h, acc[t]);
            acc[t] = wmma_bf16(a2h, b2l, acc[t]);
            acc[t] = wmma_bf16(a2l, b2h, acc[t]);
            acc[t] = wmma_bf16(a2l, b2l, acc[t]);
        }
        __syncthreads();
    }

    const int nLoc = lane & 15;
    const int mh   = lane >> 4;
#pragma unroll
    for (int v = 0; v < 8; ++v) {
        const int m = mBase + wave * 16 + mh * 8 + v;
        float* hrow = Hout + (size_t)m * HDIM + f0;
#pragma unroll
        for (int t = 0; t < 4; ++t) {
            float r = acc[t][v] + b[f0 + t * 16 + nLoc];
            if (relu) r = fmaxf(r, 0.0f);
            hrow[t * 16 + nLoc] = r;
        }
    }
}

// ---------------------------------------------------------------------------
// y[i,:] = h2[i,:] @ Wd + bd    (OUT=3, VALU path — trivially small)
// ---------------------------------------------------------------------------
__global__ __launch_bounds__(256)
void y_kernel(float* __restrict__ y, const float* __restrict__ h2,
              const float* __restrict__ Wd, const float* __restrict__ bd) {
    const int i = blockIdx.x * 256 + threadIdx.x;
    if (i >= NNODES) return;
    float s0 = bd[0], s1 = bd[1], s2 = bd[2];
    const float* hr = h2 + (size_t)i * HDIM;
#pragma unroll 8
    for (int k = 0; k < HDIM; ++k) {
        const float h = hr[k];
        s0 += h * Wd[k * 3 + 0];
        s1 += h * Wd[k * 3 + 1];
        s2 += h * Wd[k * 3 + 2];
    }
    y[i * 3 + 0] = s0;
    y[i * 3 + 1] = s1;
    y[i * 3 + 2] = s2;
}

// ---------------------------------------------------------------------------
// out[i,j] = ||y_i - y_j||  (store-bandwidth bound: 256 MB)
// ---------------------------------------------------------------------------
__global__ __launch_bounds__(256)
void cdist_kernel(float* __restrict__ out, const float* __restrict__ y) {
    const int i = blockIdx.x;
    const int j = blockIdx.y * 256 + threadIdx.x;
    const float dx = y[i * 3 + 0] - y[j * 3 + 0];
    const float dy = y[i * 3 + 1] - y[j * 3 + 1];
    const float dz = y[i * 3 + 2] - y[j * 3 + 2];
    const float d2 = dx * dx + dy * dy + dz * dz;
    out[(size_t)i * NNODES + j] = d2 > 0.f ? sqrtf(d2) : 0.f;
}

// ---------------------------------------------------------------------------
extern "C" void kernel_launch(void* const* d_in, const int* in_sizes, int n_in,
                              void* d_out, int out_size, void* d_ws, size_t ws_size,
                              hipStream_t stream) {
    const float* x   = (const float*)d_in[0];   // [8192,512]
    const float* adj = (const float*)d_in[1];   // [8192,8192]
    const float* Wl1 = (const float*)d_in[2];   // [512,128]
    const float* Wr1 = (const float*)d_in[3];   // [512,128]
    const float* b1  = (const float*)d_in[4];   // [128]
    const float* Wl2 = (const float*)d_in[5];   // [128,128]
    const float* Wr2 = (const float*)d_in[6];   // [128,128]
    const float* b2  = (const float*)d_in[7];   // [128]
    const float* Wd  = (const float*)d_in[8];   // [128,3]
    const float* bd  = (const float*)d_in[9];   // [3]
    float* out = (float*)d_out;                 // [8192,8192]

    // workspace layout (f32), ~28.2 MB total
    char*  ws   = (char*)d_ws;
    float* deg  = (float*)ws;                               // 8192
    float* agg1 = (float*)(ws + (64 << 10));                // 8192*512
    float* h    = agg1 + (size_t)NNODES * 512;              // 8192*128
    float* agg2 = h    + (size_t)NNODES * HDIM;             // 8192*128
    float* h2   = agg2 + (size_t)NNODES * HDIM;             // 8192*128
    float* yv   = h2   + (size_t)NNODES * HDIM;             // 8192*3

    // 1) degrees
    deg_kernel<<<NNODES, 256, 0, stream>>>(deg, adj);

    // 2) layer-1 aggregation: agg1 = (adj @ x)/deg   [8192,512]
    agg_kernel<<<dim3(512 / 128, NNODES / 128), 256, 0, stream>>>(agg1, adj, x, deg, 512);

    // 3) h = relu(agg1 @ Wl1 + x @ Wr1 + b1)         [8192,128]
    lin_kernel<<<dim3(HDIM / 64, NNODES / 128), 256, 0, stream>>>(h, agg1, x, Wl1, Wr1, b1, 512, 1);

    // 4) layer-2 aggregation: agg2 = (adj @ h)/deg   [8192,128]
    agg_kernel<<<dim3(HDIM / 128, NNODES / 128), 256, 0, stream>>>(agg2, adj, h, deg, HDIM);

    // 5) h2 = agg2 @ Wl2 + h @ Wr2 + b2              [8192,128]
    lin_kernel<<<dim3(HDIM / 64, NNODES / 128), 256, 0, stream>>>(h2, agg2, h, Wl2, Wr2, b2, HDIM, 0);

    // 6) y = h2 @ Wd + bd                            [8192,3]
    y_kernel<<<NNODES / 256, 256, 0, stream>>>(yv, h2, Wd, bd);

    // 7) out = cdist(y)                              [8192,8192]
    cdist_kernel<<<dim3(NNODES, NNODES / 256), 256, 0, stream>>>(out, yv);
}
